// ReversibleAttention_87711822119353
// MI455X (gfx1250) — compile-verified, matrix operand order checked by reading, not run on hardware
//
#include <hip/hip_runtime.h>
#include <hip/hip_bf16.h>
#include <math.h>

typedef __bf16 bf16_t;
typedef __attribute__((ext_vector_type(16))) __bf16 v16bf;
typedef __attribute__((ext_vector_type(8)))  __bf16 v8bf;
typedef __attribute__((ext_vector_type(8)))  float  v8f;

static constexpr int cB  = 4;
static constexpr int cL  = 4096;     // sequence length (2^12)
static constexpr int cD  = 1024;     // d_model
static constexpr int cH  = 8;        // heads
static constexpr int cDK = 128;      // head dim
static constexpr int cR  = 4;        // hash rounds
static constexpr int cNC = 64;       // chunks = L / BUCKET
static constexpr int cM  = cB * cL;  // 16384 GEMM rows

static constexpr float kScale   = 0.08838834764831845f; // 1/sqrt(128)
static constexpr float kNeg     = -1.0e9f;
static constexpr float kSelfPen = -100000.0f;

__device__ __forceinline__ v16bf ld_v16bf(const bf16_t* p) {
  v16bf v;
  __builtin_memcpy(&v, __builtin_assume_aligned(p, 16), sizeof(v));
  return v;
}
__device__ __forceinline__ v8bf ld_v8bf(const bf16_t* p) {
  v8bf v;
  __builtin_memcpy(&v, __builtin_assume_aligned(p, 16), sizeof(v));
  return v;
}

// ---------------------------------------------------------------------------
// K1/K6: Y = X @ W + bias.  X:(16384,1024) f32, W:(1024,1024) f32.
// mode 0: write f32 row-major (M x N)            (output projection)
// mode 1: write f32 to (B,H,L,DK) head layout    (q)
// mode 2: write bf16 to (B,H,L,DK) head layout   (v)
// 128x128 block tile, K-step 32, bf16 WMMA, double-buffered LDS.
// ---------------------------------------------------------------------------
__global__ __launch_bounds__(256) void gemm_bias_wmma(
    const float* __restrict__ X, const float* __restrict__ W,
    const float* __restrict__ bias, float* __restrict__ outF,
    bf16_t* __restrict__ outB, int mode)
{
  constexpr int AS = 40;  // padded bf16 row stride (80B = 20 dwords, conflict-free)
  __shared__ bf16_t As[2][128 * AS];
  __shared__ bf16_t Bt[2][128 * AS];  // B staged transposed: Bt[n][k]

  const int tid  = threadIdx.x;
  const int wave = tid >> 5;
  const int lane = tid & 31;
  const int mrow = lane & 15;
  const int kh   = lane >> 4;
  const int rowBase = blockIdx.y * 128;
  const int colBase = blockIdx.x * 128;
  const int wr = (wave >> 1) * 32;   // 32-row slab per wave pair
  const int wc = (wave & 1) * 64;    // 64-col slab

  v8f acc[2][4] = {};

  // stage tile kt = 0
  {
#pragma unroll
    for (int it = 0; it < 4; ++it) {
      int idx = tid + it * 256;
      int r = idx >> 3;
      int c = (idx & 7) * 4;
      float4 f = *(const float4*)(X + (size_t)(rowBase + r) * cD + c);
      bf16_t* d = &As[0][r * AS + c];
      d[0] = (bf16_t)f.x; d[1] = (bf16_t)f.y; d[2] = (bf16_t)f.z; d[3] = (bf16_t)f.w;
    }
#pragma unroll
    for (int it = 0; it < 4; ++it) {
      int idx = tid + it * 256;
      int kr = idx >> 5;
      int c = (idx & 31) * 4;
      float4 f = *(const float4*)(W + (size_t)kr * cD + colBase + c);
      Bt[0][(c + 0) * AS + kr] = (bf16_t)f.x;
      Bt[0][(c + 1) * AS + kr] = (bf16_t)f.y;
      Bt[0][(c + 2) * AS + kr] = (bf16_t)f.z;
      Bt[0][(c + 3) * AS + kr] = (bf16_t)f.w;
    }
  }
  __syncthreads();

  for (int kt = 0; kt < cD / 32; ++kt) {
    const int buf = kt & 1;
    if (kt + 1 < cD / 32) {
      const int nk = kt + 1, nb = buf ^ 1;
#pragma unroll
      for (int it = 0; it < 4; ++it) {
        int idx = tid + it * 256;
        int r = idx >> 3;
        int c = (idx & 7) * 4;
        float4 f = *(const float4*)(X + (size_t)(rowBase + r) * cD + nk * 32 + c);
        bf16_t* d = &As[nb][r * AS + c];
        d[0] = (bf16_t)f.x; d[1] = (bf16_t)f.y; d[2] = (bf16_t)f.z; d[3] = (bf16_t)f.w;
      }
#pragma unroll
      for (int it = 0; it < 4; ++it) {
        int idx = tid + it * 256;
        int kr = idx >> 5;
        int c = (idx & 31) * 4;
        float4 f = *(const float4*)(W + (size_t)(nk * 32 + kr) * cD + colBase + c);
        Bt[nb][(c + 0) * AS + kr] = (bf16_t)f.x;
        Bt[nb][(c + 1) * AS + kr] = (bf16_t)f.y;
        Bt[nb][(c + 2) * AS + kr] = (bf16_t)f.z;
        Bt[nb][(c + 3) * AS + kr] = (bf16_t)f.w;
      }
      if (kt + 2 < cD / 32) {  // global_prefetch_b8 of the k+2 tiles
        __builtin_prefetch(X + (size_t)(rowBase + (tid >> 1)) * cD + (kt + 2) * 32, 0, 1);
        __builtin_prefetch(W + (size_t)((kt + 2) * 32 + (tid >> 3)) * cD + colBase, 0, 1);
      }
    }
    // A fragment: lane holds row (lane&15), contiguous 16 K values, half by lane>>4
    v16bf a0 = ld_v16bf(&As[buf][(wr + mrow) * AS + kh * 16]);
    v16bf a1 = ld_v16bf(&As[buf][(wr + 16 + mrow) * AS + kh * 16]);
#pragma unroll
    for (int nt = 0; nt < 4; ++nt) {
      v16bf b = ld_v16bf(&Bt[buf][(wc + nt * 16 + mrow) * AS + kh * 16]);
      acc[0][nt] = __builtin_amdgcn_wmma_f32_16x16x32_bf16(
          false, a0, false, b, (short)0, acc[0][nt], false, false);
      acc[1][nt] = __builtin_amdgcn_wmma_f32_16x16x32_bf16(
          false, a1, false, b, (short)0, acc[1][nt], false, false);
    }
    __syncthreads();
  }

  // epilogue: C layout — VGPR vr -> row vr + 8*(lane>=16), col = lane&15
#pragma unroll
  for (int mi = 0; mi < 2; ++mi) {
#pragma unroll
    for (int nt = 0; nt < 4; ++nt) {
#pragma unroll
      for (int vr = 0; vr < 8; ++vr) {
        int row = rowBase + wr + mi * 16 + vr + 8 * kh;
        int col = colBase + wc + nt * 16 + mrow;
        float val = acc[mi][nt][vr] + bias[col];
        if (mode == 0) {
          outF[(size_t)row * cD + col] = val;
        } else {
          int b = row >> 12, l = row & (cL - 1);
          int h = col >> 7,  d = col & (cDK - 1);
          size_t qi = (((size_t)(b * cH + h)) * cL + l) * cDK + d;
          if (mode == 1) outF[qi] = val;
          else           outB[qi] = (bf16_t)val;
        }
      }
    }
  }
}

// ---------------------------------------------------------------------------
// K2: per-(b,h,l) reciprocal norm of q row: 1/(||q|| + 1e-9)
// ---------------------------------------------------------------------------
__global__ void rnorm_kernel(const float* __restrict__ q, float* __restrict__ rn)
{
  int i = blockIdx.x * blockDim.x + threadIdx.x;
  if (i >= cB * cH * cL) return;
  const float* r = q + (size_t)i * cDK;
  float s = 0.f;
#pragma unroll 16
  for (int d = 0; d < cDK; ++d) s = fmaf(r[d], r[d], s);
  rn[i] = 1.0f / (sqrtf(s) + 1e-9f);
}

// ---------------------------------------------------------------------------
// K3: LSH hash.  proj[n] = q . rot[:,r,n];  hash = argmax([proj, -proj])
// ---------------------------------------------------------------------------
__global__ void hash_kernel(const float* __restrict__ q,
                            const float* __restrict__ rot,
                            int* __restrict__ hashes)
{
  int i = blockIdx.x * blockDim.x + threadIdx.x;  // (B,H,R,L) flat
  if (i >= cB * cH * cR * cL) return;
  int l  = i & (cL - 1);
  int r  = (i >> 12) & (cR - 1);
  int bh = i >> 14;
  const float* qr = q + ((size_t)bh * cL + l) * cDK;
  float acc[32];
#pragma unroll
  for (int n = 0; n < 32; ++n) acc[n] = 0.f;
  for (int d = 0; d < cDK; ++d) {
    float qd = qr[d];
    const float* rp = rot + ((size_t)d * cR + r) * 32;
#pragma unroll
    for (int n = 0; n < 32; ++n) acc[n] = fmaf(qd, rp[n], acc[n]);
  }
  float best = acc[0];
  int bi = 0;
#pragma unroll
  for (int n = 1; n < 32; ++n)
    if (acc[n] > best) { best = acc[n]; bi = n; }
#pragma unroll
  for (int n = 0; n < 32; ++n)
    if (-acc[n] > best) { best = -acc[n]; bi = 32 + n; }
  hashes[i] = bi;
}

// ---------------------------------------------------------------------------
// K4: stable counting sort per (b,h,r): reproduces argsort(hash*L + pos).
// One block (64 threads, one per bucket) handles 4096 elements.
// ---------------------------------------------------------------------------
__global__ __launch_bounds__(64) void bucket_sort_kernel(
    const int* __restrict__ hashes, int* __restrict__ sidx)
{
  __shared__ int cnt[cNC + 1];
  const int bhr = blockIdx.x;
  const int* hp = hashes + (size_t)bhr * cL;
  int* sp = sidx + (size_t)bhr * cL;
  const int t = threadIdx.x;  // bucket id 0..63
  int c = 0;
  for (int l = 0; l < cL; ++l) c += (hp[l] == t) ? 1 : 0;
  cnt[t + 1] = c;
  __syncthreads();
  if (t == 0) {
    cnt[0] = 0;
    for (int n = 1; n <= cNC; ++n) cnt[n] += cnt[n - 1];
  }
  __syncthreads();
  int off = cnt[t];
  for (int l = 0; l < cL; ++l)
    if (hp[l] == t) sp[off++] = l;  // pos ascending within bucket -> stable
}

// ---------------------------------------------------------------------------
// K5: bucketed attention.  One block per (chunk, b*h*r).
//   S = Q Klb^T / sqrt(dk) -> mask -> softmax -> O = P Vlb, LSE.
// O and LSE are scattered directly to original positions (undoes the sort).
// Dynamic LDS: Q(64x136 bf16) K(128x136) Vt(128x136) P(64x136) S(64x132 f32)
// ---------------------------------------------------------------------------
__global__ __launch_bounds__(256) void lsh_attn_wmma(
    const float* __restrict__ qg, const bf16_t* __restrict__ vg,
    const float* __restrict__ rn, const int* __restrict__ sidx,
    const unsigned char* __restrict__ maskg,
    bf16_t* __restrict__ obuf, float* __restrict__ lsebuf)
{
  constexpr int LSB = 136;  // bf16 row stride (272B, conflict-free)
  constexpr int LSF = 132;  // f32 row stride
  extern __shared__ char dsm[];
  bf16_t* Qs = (bf16_t*)dsm;
  bf16_t* Ks = Qs + 64 * LSB;
  bf16_t* Vt = Ks + 128 * LSB;   // transposed: Vt[d][j]
  bf16_t* Ps = Vt + 128 * LSB;
  float*  Ss = (float*)(Ps + 64 * LSB);

  __shared__ int           kidx[128];
  __shared__ unsigned char kmsk[128];
  __shared__ float         red[64][4];
  __shared__ float         rs[64];

  const int c    = blockIdx.x;
  const int bhr  = blockIdx.y;          // ((b*H + h)*R + r)
  const int b    = bhr >> 5;
  const int bh   = bhr >> 2;            // b*H + h
  const int tid  = threadIdx.x;
  const int wave = tid >> 5, lane = tid & 31;
  const int mrow = lane & 15, kh = lane >> 4;
  const int wm = (wave >> 1) * 16;      // 16-row tile per wave pair
  const int wn = (wave & 1) * 64;       // 64-col slab
  const int pc = (c == 0) ? (cNC - 1) : (c - 1);  // look-back = roll(.,1)
  const int* sp = sidx + (size_t)bhr * cL;

  if (tid < 128) {  // keys: j<64 prev chunk, j>=64 current (queries = 64..127)
    int sl = (tid < 64) ? (pc * 64 + tid) : (c * 64 + (tid - 64));
    int p = sp[sl];
    kidx[tid] = p;
    kmsk[tid] = maskg[(size_t)b * cL + p];
  }
  __syncthreads();

  // stage Q (gathered, unnormalized)
#pragma unroll
  for (int it = 0; it < 8; ++it) {
    int idx = tid + it * 256;
    int row = idx >> 5;
    int cc  = (idx & 31) * 4;
    int p = kidx[64 + row];
    float4 f = *(const float4*)(qg + (((size_t)bh * cL + p) << 7) + cc);
    bf16_t* d = &Qs[row * LSB + cc];
    d[0] = (bf16_t)f.x; d[1] = (bf16_t)f.y; d[2] = (bf16_t)f.z; d[3] = (bf16_t)f.w;
  }
  // stage K = q * rnorm (gathered, normalized)
#pragma unroll
  for (int it = 0; it < 16; ++it) {
    int idx = tid + it * 256;
    int row = idx >> 5;
    int cc  = (idx & 31) * 4;
    int p = kidx[row];
    float s = rn[(size_t)bh * cL + p];
    float4 f = *(const float4*)(qg + (((size_t)bh * cL + p) << 7) + cc);
    bf16_t* d = &Ks[row * LSB + cc];
    d[0] = (bf16_t)(f.x * s); d[1] = (bf16_t)(f.y * s);
    d[2] = (bf16_t)(f.z * s); d[3] = (bf16_t)(f.w * s);
  }
  // stage V transposed (Vt[d][j]) so P@V B-fragments read contiguously
#pragma unroll
  for (int it = 0; it < 8; ++it) {
    int idx = tid + it * 256;
    int row = idx >> 4;           // key j
    int cc  = (idx & 15) * 8;     // d base
    int p = kidx[row];
    v8bf vv = ld_v8bf(vg + (((size_t)bh * cL + p) << 7) + cc);
#pragma unroll
    for (int e = 0; e < 8; ++e) Vt[(cc + e) * LSB + row] = vv[e];
  }
  __syncthreads();

  // S = Q @ K^T * scale  (M=64, N=128, K=128): 16 WMMA per wave
  {
    v8f sa[4] = {};
#pragma unroll
    for (int kt = 0; kt < 4; ++kt) {
      v16bf a = ld_v16bf(&Qs[(wm + mrow) * LSB + kt * 32 + kh * 16]);
#pragma unroll
      for (int nt = 0; nt < 4; ++nt) {
        v16bf bb = ld_v16bf(&Ks[(wn + nt * 16 + mrow) * LSB + kt * 32 + kh * 16]);
        sa[nt] = __builtin_amdgcn_wmma_f32_16x16x32_bf16(
            false, a, false, bb, (short)0, sa[nt], false, false);
      }
    }
#pragma unroll
    for (int nt = 0; nt < 4; ++nt)
#pragma unroll
      for (int vr = 0; vr < 8; ++vr)
        Ss[(wm + vr + 8 * kh) * LSF + wn + nt * 16 + mrow] = sa[nt][vr] * kScale;
  }
  __syncthreads();

  // mask + softmax: 4 threads per row, 32 cols each
  {
    const int row = tid >> 2, part = tid & 3;
    const int qp = kidx[64 + row];
    float mx = -3.0e38f;
    for (int j = part * 32; j < part * 32 + 32; ++j) {
      float s = Ss[row * LSF + j];
      int kp = kidx[j];
      if (!kmsk[j]) s = kNeg;
      if (kp > qp)  s = kNeg;
      if (kp == qp) s = kSelfPen;
      Ss[row * LSF + j] = s;
      mx = fmaxf(mx, s);
    }
    red[row][part] = mx;
    __syncthreads();
    float m = fmaxf(fmaxf(red[row][0], red[row][1]),
                    fmaxf(red[row][2], red[row][3]));
    __syncthreads();
    float sum = 0.f;
    for (int j = part * 32; j < part * 32 + 32; ++j) {
      float e = __expf(Ss[row * LSF + j] - m);
      Ps[row * LSB + j] = (bf16_t)e;
      sum += e;
    }
    red[row][part] = sum;
    __syncthreads();
    float tot = red[row][0] + red[row][1] + red[row][2] + red[row][3];
    if (part == 0) {
      rs[row] = 1.0f / tot;
      lsebuf[(size_t)bhr * cL + qp] = m + logf(tot);  // scatter (unsort)
    }
  }
  __syncthreads();

  // O = P @ V  (M=64, N=128, K=128): 16 WMMA per wave; scale by 1/sum, scatter
  {
    v8f oa[4] = {};
#pragma unroll
    for (int kt = 0; kt < 4; ++kt) {
      v16bf a = ld_v16bf(&Ps[(wm + mrow) * LSB + kt * 32 + kh * 16]);
#pragma unroll
      for (int nt = 0; nt < 4; ++nt) {
        v16bf bb = ld_v16bf(&Vt[(wn + nt * 16 + mrow) * LSB + kt * 32 + kh * 16]);
        oa[nt] = __builtin_amdgcn_wmma_f32_16x16x32_bf16(
            false, a, false, bb, (short)0, oa[nt], false, false);
      }
    }
#pragma unroll
    for (int nt = 0; nt < 4; ++nt)
#pragma unroll
      for (int vr = 0; vr < 8; ++vr) {
        int row = wm + vr + 8 * kh;
        int d   = wn + nt * 16 + mrow;
        float val = oa[nt][vr] * rs[row];
        int p = kidx[64 + row];
        obuf[(((size_t)bhr * cL + p) << 7) + d] = (bf16_t)val;  // scatter
      }
  }
}

// ---------------------------------------------------------------------------
// K7: softmax over rounds on LSE, weighted sum of O -> context (B,L,D_MODEL)
// ---------------------------------------------------------------------------
__global__ void combine_rounds(const bf16_t* __restrict__ obuf,
                               const float* __restrict__ lsebuf,
                               float* __restrict__ ctx)
{
  size_t i = (size_t)blockIdx.x * 256 + threadIdx.x;  // (B,H,L,DK) flat
  int d = (int)(i & (cDK - 1));
  size_t bhl = i >> 7;
  int l  = (int)(bhl & (cL - 1));
  int bh = (int)(bhl >> 12);
  float ls[cR];
  float m = -3.0e38f;
#pragma unroll
  for (int r = 0; r < cR; ++r) {
    ls[r] = lsebuf[((size_t)(bh * cR + r)) * cL + l];
    m = fmaxf(m, ls[r]);
  }
  float sum = 0.f;
#pragma unroll
  for (int r = 0; r < cR; ++r) { ls[r] = __expf(ls[r] - m); sum += ls[r]; }
  float inv = 1.0f / sum;
  float o = 0.f;
#pragma unroll
  for (int r = 0; r < cR; ++r)
    o += ls[r] * inv *
         (float)obuf[((((size_t)(bh * cR + r)) * cL + l) << 7) + d];
  int b = bh >> 3, h = bh & (cH - 1);
  ctx[((size_t)(b * cL + l)) * cD + h * cDK + d] = o;
}

// ---------------------------------------------------------------------------
// K8: y1 = x1 + LayerNorm(a)*gamma + beta ; y2 = x2 (tuple output, flat)
// ---------------------------------------------------------------------------
__global__ __launch_bounds__(256) void ln_residual(
    const float* __restrict__ a, const float* __restrict__ x1,
    const float* __restrict__ x2, const float* __restrict__ gamma,
    const float* __restrict__ beta, float* __restrict__ out)
{
  __shared__ float r1[256], r2[256];
  const int row = blockIdx.x;   // B*L rows
  const int tid = threadIdx.x;
  const float* ar = a + (size_t)row * cD;
  float v[4], s = 0.f, s2 = 0.f;
#pragma unroll
  for (int it = 0; it < 4; ++it) {
    v[it] = ar[tid + it * 256];
    s += v[it]; s2 += v[it] * v[it];
  }
  r1[tid] = s; r2[tid] = s2;
  __syncthreads();
  for (int off = 128; off > 0; off >>= 1) {
    if (tid < off) { r1[tid] += r1[tid + off]; r2[tid] += r2[tid + off]; }
    __syncthreads();
  }
  float mu   = r1[0] * (1.0f / cD);
  float var  = r2[0] * (1.0f / cD) - mu * mu;
  float rstd = rsqrtf(var + 1e-5f);
  const size_t base = (size_t)row * cD;
  float* y2 = out + (size_t)cM * cD;
#pragma unroll
  for (int it = 0; it < 4; ++it) {
    int col = tid + it * 256;
    float ln = (v[it] - mu) * rstd * gamma[col] + beta[col];
    out[base + col] = x1[base + col] + ln;
    y2[base + col]  = x2[base + col];
  }
}

// ---------------------------------------------------------------------------
extern "C" void kernel_launch(void* const* d_in, const int* in_sizes, int n_in,
                              void* d_out, int out_size, void* d_ws, size_t ws_size,
                              hipStream_t stream)
{
  (void)in_sizes; (void)n_in; (void)out_size; (void)ws_size;
  const float* x1    = (const float*)d_in[0];
  const float* x2    = (const float*)d_in[1];
  const float* wq    = (const float*)d_in[2];
  const float* bq    = (const float*)d_in[3];
  const float* wv    = (const float*)d_in[4];
  const float* bv    = (const float*)d_in[5];
  const float* wo    = (const float*)d_in[6];
  const float* bo    = (const float*)d_in[7];
  const float* gamma = (const float*)d_in[8];
  const float* beta  = (const float*)d_in[9];
  const float* rot   = (const float*)d_in[10];
  const unsigned char* mask = (const unsigned char*)d_in[11];
  float* out = (float*)d_out;

  char* ws = (char*)d_ws;
  size_t off = 0;
  auto wsAlloc = [&](size_t bytes) -> char* {
    char* p = ws + off;
    off += (bytes + 255) & ~(size_t)255;
    return p;
  };
  float*  qf   = (float*) wsAlloc((size_t)cB * cH * cL * cDK * 4);        // 64 MB
  bf16_t* vb   = (bf16_t*)wsAlloc((size_t)cB * cH * cL * cDK * 2);        // 32 MB
  float*  rn   = (float*) wsAlloc((size_t)cB * cH * cL * 4);              // 0.5 MB
  int*    hsh  = (int*)   wsAlloc((size_t)cB * cH * cR * cL * 4);         // 2 MB
  int*    sidx = (int*)   wsAlloc((size_t)cB * cH * cR * cL * 4);         // 2 MB
  float*  lse  = (float*) wsAlloc((size_t)cB * cH * cR * cL * 4);         // 2 MB
  bf16_t* ob   = (bf16_t*)wsAlloc((size_t)cB * cH * cR * cL * cDK * 2);   // 128 MB
  float*  ctx  = (float*) wsAlloc((size_t)cB * cL * cD * 4);              // 64 MB
  float*  am   = (float*) wsAlloc((size_t)cB * cL * cD * 4);              // 64 MB

  const dim3 gGemm(cD / 128, cM / 128);  // (8, 128)

  gemm_bias_wmma<<<gGemm, 256, 0, stream>>>(x2, wq, bq, qf, nullptr, 1);
  gemm_bias_wmma<<<gGemm, 256, 0, stream>>>(x2, wv, bv, nullptr, vb, 2);
  rnorm_kernel<<<(cB * cH * cL) / 256, 256, 0, stream>>>(qf, rn);
  hash_kernel<<<(cB * cH * cR * cL) / 256, 256, 0, stream>>>(qf, rot, hsh);
  bucket_sort_kernel<<<cB * cH * cR, 64, 0, stream>>>(hsh, sidx);

  constexpr size_t kAttnLds =
      (size_t)(64 * 136 + 128 * 136 + 128 * 136 + 64 * 136) * 2 + 64 * 132 * 4;
  static_assert(kAttnLds == 138240, "attention LDS budget");
  (void)hipFuncSetAttribute((const void*)lsh_attn_wmma,
                            hipFuncAttributeMaxDynamicSharedMemorySize,
                            (int)kAttnLds);
  const dim3 gAttn(cNC, cB * cH * cR);   // (64, 128)
  lsh_attn_wmma<<<gAttn, 256, kAttnLds, stream>>>(qf, vb, rn, sidx, mask, ob, lse);

  combine_rounds<<<(cB * cH * cL * cDK) / 256, 256, 0, stream>>>(ob, lse, ctx);
  gemm_bias_wmma<<<gGemm, 256, 0, stream>>>(ctx, wo, bo, am, nullptr, 0);
  ln_residual<<<cM, 256, 0, stream>>>(am, x1, x2, gamma, beta, out);
}
